// PaLMLayer_72705206387195
// MI455X (gfx1250) — matrix operand order repaired and checked by reading, not compiled, not run on hardware
//
#include <hip/hip_runtime.h>
#include <cstdint>

#define DIMN  2048
#define SEQ   2048
#define BSZ   2
#define NH    32
#define DH    64
#define ROWS  (BSZ*SEQ)      // 4096
#define NQKV  (DIMN+DH)      // 2112 (logical)
#define NQKVP 2176           // padded to a multiple of 128 -> no tile guards
#define FFN   16384          // w_ff1 output cols
#define FFH   8192           // half (u / gate)

typedef unsigned short u16;
typedef __attribute__((ext_vector_type(16))) __bf16 v16bf;
typedef __attribute__((ext_vector_type(8)))  float  v8f;

union FragAB { v16bf v; u16 u[16]; uint4 q[2]; };
union FragC  { v8f v; float f[8]; };

__device__ __forceinline__ u16 f2bf(float f) {
  unsigned int x = __float_as_uint(f);
  x += 0x7fffu + ((x >> 16) & 1u);        // round-to-nearest-even
  return (u16)(x >> 16);
}
__device__ __forceinline__ float lane_bcast(float v, int srclane) {
  return __int_as_float(__builtin_amdgcn_ds_bpermute(srclane << 2, __float_as_int(v)));
}

// --------------------------------------------------------- weight f32 -> bf16^T
// Reads w[K x N] (f32, row-major), writes wT[Npad x K] (bf16).  LDS-tiled so
// both the global read (along n) and write (along k) are fully coalesced.
// Rows n in [N, Npad) are zero-filled.
__global__ __launch_bounds__(256) void cvt_transpose_kernel(
    const float* __restrict__ s, u16* __restrict__ d, int K, int N, int Npad) {
  __shared__ u16 tile[64][65];            // [n][k], padded to dodge conflicts
  const int k0 = blockIdx.y * 64;
  const int n0 = blockIdx.x * 64;
#pragma unroll
  for (int i = 0; i < 16; i++) {
    const int idx = threadIdx.x + i * 256;   // 0..4095
    const int r = idx >> 6;                  // k offset 0..63
    const int c = idx & 63;                  // n offset 0..63
    const int n = n0 + c;
    tile[c][r] = (n < N) ? f2bf(s[(size_t)(k0 + r) * N + n]) : (u16)0;
  }
  __syncthreads();
#pragma unroll
  for (int i = 0; i < 16; i++) {
    const int idx = threadIdx.x + i * 256;
    const int r = idx >> 6;                  // n offset
    const int c = idx & 63;                  // k offset
    d[(size_t)(n0 + r) * K + k0 + c] = tile[r][c];
  }
  (void)Npad;  // grid is sized Npad/64; pad rows get zeros via the read guard
}

// ---------------------------------------------------------------- LayerNorm
__global__ __launch_bounds__(256) void ln_kernel(
    const float* __restrict__ x, const float* __restrict__ w,
    const float* __restrict__ b, u16* __restrict__ xn) {
  const int row = blockIdx.x;
  const int tid = threadIdx.x;
  const float* xr = x + (size_t)row * DIMN;
  float v[8], s = 0.f, s2 = 0.f;
#pragma unroll
  for (int i = 0; i < 8; i++) {
    v[i] = xr[tid + i * 256];
    s += v[i]; s2 += v[i] * v[i];
  }
  __shared__ float rs[256], rs2[256];
  rs[tid] = s; rs2[tid] = s2;
  __syncthreads();
  for (int st = 128; st > 0; st >>= 1) {
    if (tid < st) { rs[tid] += rs[tid + st]; rs2[tid] += rs2[tid + st]; }
    __syncthreads();
  }
  const float mu  = rs[0] * (1.f / DIMN);
  const float var = rs2[0] * (1.f / DIMN) - mu * mu;
  const float inv = rsqrtf(var + 1e-5f);
#pragma unroll
  for (int i = 0; i < 8; i++) {
    const int col = tid + i * 256;
    xn[(size_t)row * DIMN + col] = f2bf((v[i] - mu) * inv * w[col] + b[col]);
  }
}

// ---------------------------------------------------------------- generic WMMA GEMM
// C[M x N] = A[M x K](bf16,row) @ B, with B given pre-transposed: BT[N x K].
// Block tile 128x128, 8 waves each own 64x32 = 4x2 wmma_f32_16x16x32_bf16.
// Both A and BT stage with 2 b128 loads + 2 b128 LDS stores per thread.
// Double-buffered LDS; all dims multiples of the tile -> no guards.
// MODE 1: bf16 store, cols<DIMN scaled 0.125 (q-scale for qkv)
// MODE 2: f32 out = resid + acc     MODE 3: f32 out += acc
#define SAS 40
#define SBS 40

template <int MODE>
__global__ __launch_bounds__(256) void wmma_gemm_kernel(
    const u16* __restrict__ A, int lda,
    const u16* __restrict__ BT, int ldbt,
    int K,
    u16* __restrict__ obf, float* __restrict__ of32,
    const float* __restrict__ resid, int ldc) {
  __shared__ __attribute__((aligned(16))) u16 sA [2][128 * SAS];
  __shared__ __attribute__((aligned(16))) u16 sBt[2][128 * SBS];
  const int tid  = threadIdx.x;
  const int lane = tid & 31, wave = tid >> 5;
  const int ln = lane & 15, hb = lane >> 4;
  const int kh8 = hb * 8, kh16 = hb * 16;
  const int wm = wave & 1, wn = wave >> 1;
  const int blockM = blockIdx.y * 128;
  const int blockN = blockIdx.x * 128;

  FragC acc[4][2];
#pragma unroll
  for (int mi = 0; mi < 4; mi++)
#pragma unroll
    for (int ni = 0; ni < 2; ni++)
#pragma unroll
      for (int r = 0; r < 8; r++) acc[mi][ni].f[r] = 0.f;

  const int sr = tid >> 1, sk = (tid & 1) * 16;   // staging: 128 rows x 32 k
  const u16* aSrc0 = A  + (size_t)(blockM + sr) * lda  + sk;
  const u16* bSrc0 = BT + (size_t)(blockN + sr) * ldbt + sk;

  uint4 a0, a1, b0, b1;
  auto loadTile = [&](int k0) {
    const u16* sa = aSrc0 + k0;
    a0 = ((const uint4*)sa)[0];
    a1 = ((const uint4*)sa)[1];
    const u16* sb = bSrc0 + k0;
    b0 = ((const uint4*)sb)[0];
    b1 = ((const uint4*)sb)[1];
    if (k0 + 32 < K) {                 // prefetch one tile further into L2
      __builtin_prefetch(sa + 32, 0, 0);
      __builtin_prefetch(sb + 32, 0, 0);
    }
  };
  auto storeTile = [&](int buf) {
    *(uint4*)&sA [buf][sr * SAS + sk]     = a0;
    *(uint4*)&sA [buf][sr * SAS + sk + 8] = a1;
    *(uint4*)&sBt[buf][sr * SBS + sk]     = b0;
    *(uint4*)&sBt[buf][sr * SBS + sk + 8] = b1;
  };

  loadTile(0);
  storeTile(0);
  __syncthreads();

  const int NT = K >> 5;
  for (int kt = 0; kt < NT; kt++) {
    const int cur = kt & 1;
    const bool more = (kt + 1) < NT;
    if (more) loadTile((kt + 1) << 5);   // overlap with WMMAs below

    FragAB fa[4], fb[2];
#pragma unroll
    for (int mi = 0; mi < 4; mi++) {     // A frag: contiguous K at kh16
      const u16* s = &sA[cur][(wm * 64 + mi * 16 + ln) * SAS + kh16];
      fa[mi].q[0] = *(const uint4*)(s);
      fa[mi].q[1] = *(const uint4*)(s + 8);
    }
#pragma unroll
    for (int ni = 0; ni < 2; ni++) {     // B frag: contiguous K at kh16
      const u16* s = &sBt[cur][(wn * 32 + ni * 16 + ln) * SBS + kh16];
      fb[ni].q[0] = *(const uint4*)(s);
      fb[ni].q[1] = *(const uint4*)(s + 8);
    }
#pragma unroll
    for (int mi = 0; mi < 4; mi++)
#pragma unroll
      for (int ni = 0; ni < 2; ni++)
        acc[mi][ni].v = __builtin_amdgcn_wmma_f32_16x16x32_bf16(
            false, fa[mi].v, false, fb[ni].v, (short)0, acc[mi][ni].v, false, false);

    if (more) {
      storeTile(1 - cur);
      __syncthreads();
    }
  }

#pragma unroll
  for (int mi = 0; mi < 4; mi++)
#pragma unroll
    for (int ni = 0; ni < 2; ni++) {
      const int col = blockN + wn * 32 + ni * 16 + ln;
#pragma unroll
      for (int r = 0; r < 8; r++) {
        const int row = blockM + wm * 64 + mi * 16 + kh8 + r;
        const float v = acc[mi][ni].f[r];
        const size_t idx = (size_t)row * ldc + col;
        if (MODE == 1)      obf[idx] = f2bf(col < DIMN ? v * 0.125f : v);
        else if (MODE == 2) of32[idx] = resid[idx] + v;
        else                of32[idx] += v;
      }
    }
}

// ---------------------------------------------------------------- fused ff1 + SwiGLU
// BT = w_ff1^T [16384 x 2048]: u rows [j, j+128), gate rows [j+FFH, ...).
// Computes both halves per block and stores silu(gate)*u as bf16, never
// materializing the 16384-wide h.
__global__ __launch_bounds__(256) void wmma_ff1_swiglu_kernel(
    const u16* __restrict__ A, const u16* __restrict__ BT, u16* __restrict__ out) {
  __shared__ __attribute__((aligned(16))) u16 sA [2][128 * SAS];
  __shared__ __attribute__((aligned(16))) u16 sBu[2][128 * SBS];
  __shared__ __attribute__((aligned(16))) u16 sBg[2][128 * SBS];
  const int tid  = threadIdx.x;
  const int lane = tid & 31, wave = tid >> 5;
  const int ln = lane & 15, hb = lane >> 4;
  const int kh8 = hb * 8, kh16 = hb * 16;
  const int wm = wave & 1, wn = wave >> 1;
  const int blockM = blockIdx.y * 128;
  const int blockN = blockIdx.x * 128;

  FragC au[4][2], ag[4][2];
#pragma unroll
  for (int mi = 0; mi < 4; mi++)
#pragma unroll
    for (int ni = 0; ni < 2; ni++)
#pragma unroll
      for (int r = 0; r < 8; r++) { au[mi][ni].f[r] = 0.f; ag[mi][ni].f[r] = 0.f; }

  const int sr = tid >> 1, sk = (tid & 1) * 16;
  const u16* aSrc0 = A  + (size_t)(blockM + sr) * DIMN + sk;
  const u16* uSrc0 = BT + (size_t)(blockN + sr) * DIMN + sk;
  const u16* gSrc0 = uSrc0 + (size_t)FFH * DIMN;

  uint4 a0, a1, u0, u1, g0, g1;
  auto loadTile = [&](int k0) {
    const u16* sa = aSrc0 + k0;
    a0 = ((const uint4*)sa)[0];
    a1 = ((const uint4*)sa)[1];
    const u16* su = uSrc0 + k0;
    u0 = ((const uint4*)su)[0];
    u1 = ((const uint4*)su)[1];
    const u16* sg = gSrc0 + k0;
    g0 = ((const uint4*)sg)[0];
    g1 = ((const uint4*)sg)[1];
    if (k0 + 32 < DIMN) {
      __builtin_prefetch(sa + 32, 0, 0);
      __builtin_prefetch(su + 32, 0, 0);
      __builtin_prefetch(sg + 32, 0, 0);
    }
  };
  auto storeTile = [&](int buf) {
    *(uint4*)&sA [buf][sr * SAS + sk]     = a0;
    *(uint4*)&sA [buf][sr * SAS + sk + 8] = a1;
    *(uint4*)&sBu[buf][sr * SBS + sk]     = u0;
    *(uint4*)&sBu[buf][sr * SBS + sk + 8] = u1;
    *(uint4*)&sBg[buf][sr * SBS + sk]     = g0;
    *(uint4*)&sBg[buf][sr * SBS + sk + 8] = g1;
  };

  loadTile(0);
  storeTile(0);
  __syncthreads();

  const int NT = DIMN >> 5;
  for (int kt = 0; kt < NT; kt++) {
    const int cur = kt & 1;
    const bool more = (kt + 1) < NT;
    if (more) loadTile((kt + 1) << 5);

    FragAB fa[4], fbu[2], fbg[2];
#pragma unroll
    for (int mi = 0; mi < 4; mi++) {
      const u16* s = &sA[cur][(wm * 64 + mi * 16 + ln) * SAS + kh16];
      fa[mi].q[0] = *(const uint4*)(s);
      fa[mi].q[1] = *(const uint4*)(s + 8);
    }
#pragma unroll
    for (int ni = 0; ni < 2; ni++) {
      const u16* su = &sBu[cur][(wn * 32 + ni * 16 + ln) * SBS + kh16];
      const u16* sg = &sBg[cur][(wn * 32 + ni * 16 + ln) * SBS + kh16];
      fbu[ni].q[0] = *(const uint4*)(su); fbu[ni].q[1] = *(const uint4*)(su + 8);
      fbg[ni].q[0] = *(const uint4*)(sg); fbg[ni].q[1] = *(const uint4*)(sg + 8);
    }
#pragma unroll
    for (int mi = 0; mi < 4; mi++)
#pragma unroll
      for (int ni = 0; ni < 2; ni++) {
        au[mi][ni].v = __builtin_amdgcn_wmma_f32_16x16x32_bf16(
            false, fa[mi].v, false, fbu[ni].v, (short)0, au[mi][ni].v, false, false);
        ag[mi][ni].v = __builtin_amdgcn_wmma_f32_16x16x32_bf16(
            false, fa[mi].v, false, fbg[ni].v, (short)0, ag[mi][ni].v, false, false);
      }

    if (more) {
      storeTile(1 - cur);
      __syncthreads();
    }
  }

#pragma unroll
  for (int mi = 0; mi < 4; mi++)
#pragma unroll
    for (int ni = 0; ni < 2; ni++) {
      const int col = blockN + wn * 32 + ni * 16 + ln;
#pragma unroll
      for (int r = 0; r < 8; r++) {
        const int row = blockM + wm * 64 + mi * 16 + kh8 + r;
        const float u = au[mi][ni].f[r];
        const float g = ag[mi][ni].f[r];
        const float act = (g / (1.f + __expf(-g))) * u;   // silu(g) * u
        out[(size_t)row * FFH + col] = f2bf(act);
      }
    }
}

// ---------------------------------------------------------------- flash attention
// One wave per (batch, head, 16-query tile).  S^T = K_tile x Q^T so the
// softmax (over keys) reduces within a lane (8 vals) + one lane^16 bpermute.
__global__ __launch_bounds__(32) void flash_attn_kernel(
    const u16* __restrict__ proj, u16* __restrict__ ctx) {
  __shared__ __attribute__((aligned(16))) u16 p_lds[16 * 32];  // P in A-frag layout
  __shared__ __attribute__((aligned(16))) u16 v_t [64 * 32];   // V^T: [d][k]
  const int lane = threadIdx.x;
  const int ln = lane & 15, hb = lane >> 4;
  const int kh8 = hb * 8, kh16 = hb * 16;
  const int gid = blockIdx.x;
  const int qt = gid & 127;
  const int h  = (gid >> 7) & 31;
  const int bb = gid >> 12;
  const int row0 = bb * SEQ + qt * 16;

  // Q^T fragments (q already scaled by 0.125 in the qkv GEMM epilogue)
  const u16* qrow = proj + (size_t)(row0 + ln) * NQKVP + h * DH;
  FragAB bq0, bq1;
  bq0.q[0] = *(const uint4*)(qrow + kh8);
  bq0.q[1] = *(const uint4*)(qrow + kh8 + 16);
  bq1.q[0] = *(const uint4*)(qrow + 32 + kh8);
  bq1.q[1] = *(const uint4*)(qrow + 48 + kh8);

  FragC o[4];
#pragma unroll
  for (int t = 0; t < 4; t++)
#pragma unroll
    for (int r = 0; r < 8; r++) o[t].f[r] = 0.f;
  float mrow = -1e30f, lrow = 0.f;

  const u16* kvb = proj + (size_t)(bb * SEQ) * NQKVP + DIMN;

  for (int kb = 0; kb < SEQ / 32; ++kb) {
    const u16* kr0 = kvb + (size_t)(kb * 32 + ln) * NQKVP;
    const u16* kr1 = kr0 + (size_t)16 * NQKVP;
    FragAB a00, a01, a10, a11;
    a00.q[0] = *(const uint4*)(kr0 + kh8);       a00.q[1] = *(const uint4*)(kr0 + kh8 + 16);
    a01.q[0] = *(const uint4*)(kr0 + 32 + kh8);  a01.q[1] = *(const uint4*)(kr0 + 48 + kh8);
    a10.q[0] = *(const uint4*)(kr1 + kh8);       a10.q[1] = *(const uint4*)(kr1 + kh8 + 16);
    a11.q[0] = *(const uint4*)(kr1 + 32 + kh8);  a11.q[1] = *(const uint4*)(kr1 + 48 + kh8);

    v8f zero = {};
    FragC s0, s1;
    s0.v = __builtin_amdgcn_wmma_f32_16x16x32_bf16(false, a00.v, false, bq0.v, (short)0, zero, false, false);
    s0.v = __builtin_amdgcn_wmma_f32_16x16x32_bf16(false, a01.v, false, bq1.v, (short)0, s0.v, false, false);
    s1.v = __builtin_amdgcn_wmma_f32_16x16x32_bf16(false, a10.v, false, bq0.v, (short)0, zero, false, false);
    s1.v = __builtin_amdgcn_wmma_f32_16x16x32_bf16(false, a11.v, false, bq1.v, (short)0, s1.v, false, false);

    { // stage V^T tile (32 keys x 64 d) into LDS
      const u16* vr = kvb + (size_t)(kb * 32 + lane) * NQKVP;
#pragma unroll
      for (int c = 0; c < 8; c++) {
        union { uint4 q; u16 u[8]; } p;
        p.q = *(const uint4*)(vr + c * 8);
#pragma unroll
        for (int j = 0; j < 8; j++) v_t[(c * 8 + j) * 32 + lane] = p.u[j];
      }
    }

    // online softmax over this 32-key block (per query column)
    float mt = -1e30f;
#pragma unroll
    for (int r = 0; r < 8; r++) mt = fmaxf(mt, fmaxf(s0.f[r], s1.f[r]));
    mt = fmaxf(mt, lane_bcast(mt, lane ^ 16));
    const float mnew  = fmaxf(mrow, mt);
    const float alpha = __expf(mrow - mnew);
    float p0[8], p1[8], psum = 0.f;
#pragma unroll
    for (int r = 0; r < 8; r++) {
      p0[r] = __expf(s0.f[r] - mnew);
      p1[r] = __expf(s1.f[r] - mnew);
      psum += p0[r] + p1[r];
    }
    psum += lane_bcast(psum, lane ^ 16);
    lrow = lrow * alpha + psum;
    mrow = mnew;

    { // store P to LDS in the A-fragment memory layout (q-major, stride 32)
      union { uint4 q; u16 u[8]; } w;
#pragma unroll
      for (int r = 0; r < 8; r++) w.u[r] = f2bf(p0[r]);
      *(uint4*)&p_lds[ln * 32 + kh8] = w.q;
#pragma unroll
      for (int r = 0; r < 8; r++) w.u[r] = f2bf(p1[r]);
      *(uint4*)&p_lds[ln * 32 + 16 + kh8] = w.q;
    }
    asm volatile("s_wait_dscnt 0" ::: "memory");  // same-wave LDS store->load

    // rescale O accumulators: row r of O is query kh8+r -> broadcast alpha
    float a8[8];
#pragma unroll
    for (int r = 0; r < 8; r++) a8[r] = lane_bcast(alpha, kh8 + r);
#pragma unroll
    for (int t = 0; t < 4; t++)
#pragma unroll
      for (int r = 0; r < 8; r++) o[t].f[r] *= a8[r];

    // O += P x V
    FragAB fp;
    const u16* ps = &p_lds[ln * 32 + kh8];
    fp.q[0] = *(const uint4*)(ps);
    fp.q[1] = *(const uint4*)(ps + 16);
#pragma unroll
    for (int t = 0; t < 4; t++) {
      FragAB fv;
      const u16* vs = &v_t[(t * 16 + ln) * 32 + kh8];
      fv.q[0] = *(const uint4*)(vs);
      fv.q[1] = *(const uint4*)(vs + 16);
      o[t].v = __builtin_amdgcn_wmma_f32_16x16x32_bf16(
          false, fp.v, false, fv.v, (short)0, o[t].v, false, false);
    }
  }

  // normalize and store context (bf16) for the attn_out GEMM
  const float inv = 1.f / lrow;
  float i8[8];
#pragma unroll
  for (int r = 0; r < 8; r++) i8[r] = lane_bcast(inv, kh8 + r);
#pragma unroll
  for (int t = 0; t < 4; t++)
#pragma unroll
    for (int r = 0; r < 8; r++)
      ctx[(size_t)(row0 + kh8 + r) * DIMN + h * DH + t * 16 + ln] = f2bf(o[t].f[r] * i8[r]);
}

// ---------------------------------------------------------------- launcher
extern "C" void kernel_launch(void* const* d_in, const int* in_sizes, int n_in,
                              void* d_out, int out_size, void* d_ws, size_t ws_size,
                              hipStream_t stream) {
  (void)in_sizes; (void)n_in; (void)out_size; (void)ws_size;
  const float* in_x  = (const float*)d_in[0];
  const float* ln_w  = (const float*)d_in[1];
  const float* ln_b  = (const float*)d_in[2];
  const float* w_qkv = (const float*)d_in[3];
  const float* w_ao  = (const float*)d_in[4];
  const float* w_ff1 = (const float*)d_in[5];
  const float* w_ff2 = (const float*)d_in[6];
  float* out = (float*)d_out;

  size_t off = 0;
  auto alloc = [&](size_t bytes) -> void* {
    void* p = (char*)d_ws + off;
    off = (off + bytes + 255) & ~(size_t)255;
    return p;
  };
  u16* xn     = (u16*)alloc((size_t)ROWS  * DIMN * 2);
  u16* wqkvT  = (u16*)alloc((size_t)NQKVP * DIMN * 2);   // [2176 x 2048], pad=0
  u16* waoT   = (u16*)alloc((size_t)DIMN  * DIMN * 2);   // [2048 x 2048]
  u16* wff1T  = (u16*)alloc((size_t)FFN   * DIMN * 2);   // [16384 x 2048]
  u16* wff2T  = (u16*)alloc((size_t)DIMN  * FFH  * 2);   // [2048 x 8192]
  u16* ffact  = (u16*)alloc((size_t)ROWS  * FFH  * 2);
  u16* proj   = (u16*)alloc((size_t)ROWS  * NQKVP * 2);  // padded stride
  u16* ctx    = (u16*)alloc((size_t)ROWS  * DIMN * 2);

  // weight conversion f32 -> bf16, transposed to BT[n][k] (coalesced both ways)
  cvt_transpose_kernel<<<dim3(NQKVP / 64, DIMN / 64), 256, 0, stream>>>(
      w_qkv, wqkvT, DIMN, NQKV, NQKVP);
  cvt_transpose_kernel<<<dim3(DIMN / 64, DIMN / 64), 256, 0, stream>>>(
      w_ao, waoT, DIMN, DIMN, DIMN);
  cvt_transpose_kernel<<<dim3(FFN / 64, DIMN / 64), 256, 0, stream>>>(
      w_ff1, wff1T, DIMN, FFN, FFN);
  cvt_transpose_kernel<<<dim3(DIMN / 64, FFH / 64), 256, 0, stream>>>(
      w_ff2, wff2T, FFH, DIMN, DIMN);

  // LayerNorm -> bf16
  ln_kernel<<<ROWS, 256, 0, stream>>>(in_x, ln_w, ln_b, xn);

  // fused ff1 + SwiGLU: ffact = silu(xn@Wg) * (xn@Wu)
  wmma_ff1_swiglu_kernel<<<dim3(FFH / 128, ROWS / 128), 256, 0, stream>>>(xn, wff1T, ffact);

  // proj = xn @ w_qkv (padded N=2176; q cols pre-scaled by DIM_HEAD^-0.5)
  wmma_gemm_kernel<1><<<dim3(NQKVP / 128, ROWS / 128), 256, 0, stream>>>(
      xn, DIMN, wqkvT, DIMN, DIMN, proj, nullptr, nullptr, NQKVP);

  // flash attention -> ctx
  flash_attn_kernel<<<BSZ * NH * (SEQ / 16), 32, 0, stream>>>(proj, ctx);

  // out = in_x + ffact @ w_ff2
  wmma_gemm_kernel<2><<<dim3(DIMN / 128, ROWS / 128), 256, 0, stream>>>(
      ffact, FFH, wff2T, FFH, FFH, nullptr, out, in_x, DIMN);

  // out += ctx @ w_attn_out
  wmma_gemm_kernel<3><<<dim3(DIMN / 128, ROWS / 128), 256, 0, stream>>>(
      ctx, DIMN, waoT, DIMN, DIMN, nullptr, out, nullptr, DIMN);
}
